// Devign_simplify_85341000171648
// MI455X (gfx1250) — matrix operand order repaired and checked by reading, not compiled, not run on hardware
//
#include <hip/hip_runtime.h>
#include <hip/hip_fp16.h>
#include <math.h>

#define IN_DIM   100
#define OUT_DIM  200
#define GATE_DIM 600
#define NUM_STEPS 4
#define KPAD     224   // 7 * 32
#define MPAD     208   // 13 * 16 (message cols)
#define GPAD     608   // 38 * 16 (gate cols)

typedef __attribute__((ext_vector_type(16))) _Float16 v16h;
typedef __attribute__((ext_vector_type(8)))  _Float16 v8h;
typedef __attribute__((ext_vector_type(8)))  float    v8f;
typedef __attribute__((__vector_size__(16))) int      i32x4;

#define GLOBAL_AS __attribute__((address_space(1)))
#define LDS_AS    __attribute__((address_space(3)))

__device__ __forceinline__ v8f wmma_f16(v16h a, v16h b, v8f c) {
  return __builtin_amdgcn_wmma_f32_16x16x32_f16(false, a, false, b, (short)0, c,
                                                false, false);
}

__device__ __forceinline__ v16h cat8(v8h lo, v8h hi) {
  return __builtin_shufflevector(lo, hi, 0, 1, 2, 3, 4, 5, 6, 7,
                                 8, 9, 10, 11, 12, 13, 14, 15);
}

// ---- fragment loaders: fully unconditional, 16B vector loads -------------
// A 16x32 f16 frag from LDS-staged 16xKPAD tile.
__device__ __forceinline__ v16h a_frag_lds(const _Float16* S, int k0, int lane) {
  const _Float16* p = S + (lane & 15) * KPAD + k0 + ((lane >> 4) << 3);
  return cat8(*(const v8h*)p, *(const v8h*)(p + 16));
}

// A frag from global f16 row-major [Npad x KPAD]
__device__ __forceinline__ v16h a_frag_g(const _Float16* __restrict__ G,
                                         int m0, int k0, int lane) {
  const _Float16* p = G + (size_t)(m0 + (lane & 15)) * KPAD + k0 + ((lane >> 4) << 3);
  return cat8(*(const v8h*)p, *(const v8h*)(p + 16));
}

// B 32x16 frag: weights stored "Bt" style [cols x KPAD] (K contiguous per lane)
__device__ __forceinline__ v16h b_frag_g(const _Float16* __restrict__ Bt,
                                         int n0, int k0, int lane) {
  const _Float16* p = Bt + (size_t)(n0 + (lane & 15)) * KPAD + k0 + ((lane >> 4) << 3);
  return cat8(*(const v8h*)p, *(const v8h*)(p + 16));
}

// cooperative copy of 16 rows (16*KPAD halves, contiguous) into LDS.
// Prefer the CDNA5 async Global->LDS path (ASYNCcnt-tracked, no VGPR bounce).
__device__ __forceinline__ void stage_tile(const _Float16* __restrict__ g,
                                           _Float16* s) {
#if defined(__has_builtin) && __has_builtin(__builtin_amdgcn_global_load_async_to_lds_b128)
  for (int i = threadIdx.x; i < (16 * KPAD) / 8; i += blockDim.x) {
    __builtin_amdgcn_global_load_async_to_lds_b128(
        (GLOBAL_AS i32x4*)(g + (size_t)i * 8),
        (LDS_AS i32x4*)(s + i * 8), 0, 0);
  }
#if __has_builtin(__builtin_amdgcn_s_wait_asynccnt)
  __builtin_amdgcn_s_wait_asynccnt(0);
#else
  asm volatile("s_wait_asynccnt 0x0" ::: "memory");
#endif
#else
  const uint4* gp = (const uint4*)g;
  uint4* sp = (uint4*)s;
  for (int i = threadIdx.x; i < (16 * KPAD) / 8; i += blockDim.x) sp[i] = gp[i];
#endif
}

__device__ __forceinline__ float sigmoidf_dev(float x) {
  return 1.0f / (1.0f + __expf(-x));
}

// ---- weight / bias prep (once per launch, tiny) --------------------------
__global__ void ggnn_conv_W(const float* __restrict__ W, _Float16* __restrict__ WFt) {
  int i = blockIdx.x * blockDim.x + threadIdx.x;
  if (i >= NUM_STEPS * MPAD * KPAD) return;
  int k = i % KPAD;
  int n = (i / KPAD) % MPAD;
  int s = i / (KPAD * MPAD);
  float v = (n < OUT_DIM && k < OUT_DIM)
                ? W[(size_t)s * OUT_DIM * OUT_DIM + (size_t)k * OUT_DIM + n] : 0.0f;
  WFt[i] = (_Float16)v;
}

__global__ void ggnn_conv_gw(const float* __restrict__ src, _Float16* __restrict__ dst) {
  int i = blockIdx.x * blockDim.x + threadIdx.x;
  if (i >= GPAD * KPAD) return;
  int k = i % KPAD;
  int g = i / KPAD;
  dst[i] = (_Float16)((g < GATE_DIM && k < OUT_DIM) ? src[(size_t)g * OUT_DIM + k] : 0.0f);
}

__global__ void ggnn_conv_bias(const float* __restrict__ bi, const float* __restrict__ bh,
                               float* __restrict__ biF, float* __restrict__ bhF) {
  int g = blockIdx.x * blockDim.x + threadIdx.x;
  if (g >= GPAD) return;
  biF[g] = (g < GATE_DIM) ? bi[g] : 0.0f;
  bhF[g] = (g < GATE_DIM) ? bh[g] : 0.0f;
}

// ---- activations ---------------------------------------------------------
__global__ void ggnn_init_h(const float* __restrict__ x, float* __restrict__ h,
                            _Float16* __restrict__ hF, int N, long long total) {
  long long i = (long long)blockIdx.x * blockDim.x + threadIdx.x;
  if (i >= total) return;
  int c = (int)(i % KPAD);
  long long n = i / KPAD;
  float f = (c < IN_DIM && n < N) ? x[n * IN_DIM + c] : 0.0f;
  hF[i] = (_Float16)f;
  if (c < OUT_DIM) h[n * OUT_DIM + c] = f;
}

__global__ void ggnn_zero_u32(unsigned* __restrict__ p, long long n) {
  long long i = (long long)blockIdx.x * blockDim.x + threadIdx.x;
  if (i < n) p[i] = 0u;
}

// ---- m = h @ W[i] : 16-row block; all 7 B frags preloaded for deep clause -
__global__ __launch_bounds__(256)
void ggnn_msg_gemm(const _Float16* __restrict__ hF, const _Float16* __restrict__ WFt,
                   _Float16* __restrict__ msgF) {
  __shared__ _Float16 sH[16 * KPAD] __attribute__((aligned(16)));
  int m0 = blockIdx.x * 16;
  stage_tile(hF + (size_t)m0 * KPAD, sH);
  __syncthreads();
  int wave = threadIdx.x >> 5;
  int lane = threadIdx.x & 31;
  for (int nt = wave; nt < 13; nt += 8) {
    int n0 = nt * 16;
    v16h bfr[7];
#pragma unroll
    for (int kc = 0; kc < 7; ++kc) bfr[kc] = b_frag_g(WFt, n0, kc * 32, lane);
    v8f acc = {};
#pragma unroll
    for (int kc = 0; kc < 7; ++kc)
      acc = wmma_f16(a_frag_lds(sH, kc * 32, lane), bfr[kc], acc);
    int col   = n0 + (lane & 15);
    int rbase = (lane >> 4) * 8;
#pragma unroll
    for (int v = 0; v < 8; ++v)
      msgF[(size_t)(m0 + rbase + v) * MPAD + col] = (_Float16)acc[v];
  }
}

// ---- scatter: aggrF[dst] += msgF[src] via packed f16 atomics -------------
__global__ void ggnn_scatter_add(const _Float16* __restrict__ msgF,
                                 const long long* __restrict__ src,
                                 const long long* __restrict__ dst,
                                 _Float16* __restrict__ aggrF, int E) {
  int e = blockIdx.x;
  if (e >= E) return;
  long long s = src[e];
  long long d = dst[e];
  int c = threadIdx.x;                     // half2 index
  if (c < OUT_DIM / 2) {
    const __half2* mp = (const __half2*)(msgF + s * MPAD) + c;
    __half2* ap = (__half2*)(aggrF + d * KPAD) + c;
    atomicAdd(ap, *mp);                    // global_atomic_pk_add_f16
  }
}

// ---- fused GRU with rotating B prefetch ----------------------------------
__global__ __launch_bounds__(256)
void ggnn_gru_fused(const _Float16* __restrict__ aggrF, float* __restrict__ h,
                    _Float16* __restrict__ hF,
                    const _Float16* __restrict__ wihF, const _Float16* __restrict__ whhF,
                    const float* __restrict__ biF, const float* __restrict__ bhF) {
  __shared__ _Float16 sH[16 * KPAD] __attribute__((aligned(16)));  // 7.0 KB
  __shared__ float lds_a[16 * GPAD];                               // 38.9 KB
  __shared__ float lds_hn[16 * MPAD];                              // 13.3 KB
  int m0 = blockIdx.x * 16;
  stage_tile(hF + (size_t)m0 * KPAD, sH);
  __syncthreads();
  int wave = threadIdx.x >> 5;
  int lane = threadIdx.x & 31;

  for (int nt = wave; nt < 38; nt += 8) {
    int n0 = nt * 16;
    v8f ai = {};
    v8f ah = {};
    v16h bi_c = b_frag_g(wihF, n0, 0, lane);
    v16h bh_c = b_frag_g(whhF, n0, 0, lane);
#pragma unroll
    for (int kc = 0; kc < 7; ++kc) {
      int k0 = kc * 32;
      v16h bi_n, bh_n;
      if (kc < 6) {
        bi_n = b_frag_g(wihF, n0, k0 + 32, lane);
        bh_n = b_frag_g(whhF, n0, k0 + 32, lane);
      }
      ai = wmma_f16(a_frag_g(aggrF, m0, k0, lane), bi_c, ai);
      ah = wmma_f16(a_frag_lds(sH, k0, lane),      bh_c, ah);
      if (kc < 6) { bi_c = bi_n; bh_c = bh_n; }
    }
    int col   = n0 + (lane & 15);
    int rbase = (lane >> 4) * 8;
    float bi = biF[col];
    float bh = bhF[col];
    if (nt < 25) {        // cols 0..399: r,z gates need gi+gh only
#pragma unroll
      for (int v = 0; v < 8; ++v)
        lds_a[(rbase + v) * GPAD + col] = ai[v] + ah[v] + bi + bh;
    } else {              // cols 400..607: n gate keeps i_n and h_n separate
#pragma unroll
      for (int v = 0; v < 8; ++v) {
        lds_a[(rbase + v) * GPAD + col]           = ai[v] + bi;
        lds_hn[(rbase + v) * MPAD + (col - 400)]  = ah[v] + bh;
      }
    }
  }
  __syncthreads();

  for (int idx = threadIdx.x; idx < 16 * OUT_DIM; idx += blockDim.x) {
    int rr = idx / OUT_DIM;
    int c  = idx - rr * OUT_DIM;
    int row = m0 + rr;
    float r = sigmoidf_dev(lds_a[rr * GPAD + c]);
    float z = sigmoidf_dev(lds_a[rr * GPAD + OUT_DIM + c]);
    float n = tanhf(lds_a[rr * GPAD + 2 * OUT_DIM + c] + r * lds_hn[rr * MPAD + c]);
    size_t off = (size_t)row * OUT_DIM + c;
    float hold = h[off];
    float hnew = (1.0f - z) * n + z * hold;
    h[off] = hnew;
    hF[(size_t)row * KPAD + c] = (_Float16)hnew;
  }
}

// ---- epilogue ------------------------------------------------------------
__global__ void ggnn_pool_max(const float* __restrict__ h, float* __restrict__ pooled,
                              int N) {
  __shared__ float red[256];
  int c = blockIdx.x;
  float mx = 0.0f;  // relu floor
  for (int r = threadIdx.x; r < N; r += blockDim.x)
    mx = fmaxf(mx, h[(size_t)r * OUT_DIM + c]);
  red[threadIdx.x] = mx;
  __syncthreads();
  for (int s = 128; s > 0; s >>= 1) {
    if (threadIdx.x < s) red[threadIdx.x] = fmaxf(red[threadIdx.x], red[threadIdx.x + s]);
    __syncthreads();
  }
  if (threadIdx.x == 0) pooled[c] = red[0];
}

__global__ void ggnn_classify(const float* __restrict__ pooled,
                              const float* __restrict__ cls_w,
                              const float* __restrict__ cls_b,
                              float* __restrict__ out) {
  if (threadIdx.x == 0) {
    float l0 = cls_b[0], l1 = cls_b[1];
    for (int k = 0; k < OUT_DIM; ++k) {
      float p = pooled[k];
      l0 += p * cls_w[k];
      l1 += p * cls_w[OUT_DIM + k];
    }
    float mx = fmaxf(l0, l1);
    float e0 = __expf(l0 - mx), e1 = __expf(l1 - mx);
    float inv = 1.0f / (e0 + e1);
    out[0] = e0 * inv;
    out[1] = e1 * inv;
  }
}

// ---------------------------------------------------------------------------
extern "C" void kernel_launch(void* const* d_in, const int* in_sizes, int n_in,
                              void* d_out, int out_size, void* d_ws, size_t ws_size,
                              hipStream_t stream) {
  const float* x     = (const float*)d_in[0];
  const float* W     = (const float*)d_in[1];
  const float* w_ih  = (const float*)d_in[2];
  const float* w_hh  = (const float*)d_in[3];
  const float* b_ih  = (const float*)d_in[4];
  const float* b_hh  = (const float*)d_in[5];
  const float* cls_w = (const float*)d_in[6];
  const float* cls_b = (const float*)d_in[7];
  const long long* ei = (const long long*)d_in[8];

  int N = in_sizes[0] / IN_DIM;
  int E = in_sizes[8] / 2;
  int mtiles = (N + 15) / 16;
  int Npad = mtiles * 16;        // all activation buffers row-padded: no guards

  char* p = (char*)d_ws;
  float*    h     = (float*)p;    p += (size_t)Npad * OUT_DIM * sizeof(float);
  _Float16* hF    = (_Float16*)p; p += (size_t)Npad * KPAD * sizeof(_Float16);
  _Float16* aggrF = (_Float16*)p; p += (size_t)Npad * KPAD * sizeof(_Float16);
  _Float16* msgF  = (_Float16*)p; p += (size_t)Npad * MPAD * sizeof(_Float16);
  _Float16* WFt   = (_Float16*)p; p += (size_t)NUM_STEPS * MPAD * KPAD * sizeof(_Float16);
  _Float16* wihF  = (_Float16*)p; p += (size_t)GPAD * KPAD * sizeof(_Float16);
  _Float16* whhF  = (_Float16*)p; p += (size_t)GPAD * KPAD * sizeof(_Float16);
  float*    biF   = (float*)p;    p += GPAD * sizeof(float);
  float*    bhF   = (float*)p;    p += GPAD * sizeof(float);
  float*    pooled = (float*)p;

  ggnn_conv_W<<<(NUM_STEPS * MPAD * KPAD + 255) / 256, 256, 0, stream>>>(W, WFt);
  ggnn_conv_gw<<<(GPAD * KPAD + 255) / 256, 256, 0, stream>>>(w_ih, wihF);
  ggnn_conv_gw<<<(GPAD * KPAD + 255) / 256, 256, 0, stream>>>(w_hh, whhF);
  ggnn_conv_bias<<<(GPAD + 255) / 256, 256, 0, stream>>>(b_ih, b_hh, biF, bhF);

  long long hElems = (long long)Npad * KPAD;
  ggnn_init_h<<<(int)((hElems + 255) / 256), 256, 0, stream>>>(x, h, hF, N, hElems);

  long long aggrDwords = (long long)Npad * KPAD / 2;

  for (int step = 0; step < NUM_STEPS; ++step) {
    ggnn_msg_gemm<<<mtiles, 256, 0, stream>>>(hF, WFt + (size_t)step * MPAD * KPAD,
                                              msgF);
    ggnn_zero_u32<<<(int)((aggrDwords + 255) / 256), 256, 0, stream>>>(
        (unsigned*)aggrF, aggrDwords);
    ggnn_scatter_add<<<E, 128, 0, stream>>>(msgF, ei, ei + E, aggrF, E);
    ggnn_gru_fused<<<mtiles, 256, 0, stream>>>(aggrF, h, hF, wihF, whhF, biF, bhF);
  }

  ggnn_pool_max<<<OUT_DIM, 256, 0, stream>>>(h, pooled, N);
  ggnn_classify<<<1, 32, 0, stream>>>(pooled, cls_w, cls_b, (float*)d_out);
}
